// BondPredictor_9259949490695
// MI455X (gfx1250) — compile-verified, matrix operand order checked by reading, not run on hardware
//
#include <hip/hip_runtime.h>

// =====================================================================
// BondPredictor for MI455X (gfx1250, wave32).
// All dense matmuls -> v_wmma_f32_16x16x32_bf16 (bf16 in, f32 acc).
// A/W pre-converted to bf16 (halves HBM stream); W tiles staged into
// LDS with global_load_async_to_lds_b128 + s_wait_asynccnt (CDNA5 async
// path); A fragments are two b128 loads. Vector (N,C,3) features kept
// planar: 3 contiguous (N,C) planes.
// =====================================================================

typedef __attribute__((ext_vector_type(16))) __bf16 bfx16;
typedef __attribute__((ext_vector_type(8)))  float  fx8;

#define PI_F 3.14159265358979f

#define GS(i, n)                                                         \
  for (long i = (long)blockIdx.x * blockDim.x + threadIdx.x; i < (n);    \
       i += (long)gridDim.x * blockDim.x)

// f32 -> bf16 (RNE) as raw u16
__device__ inline unsigned short f2bfu(float f) {
  unsigned u = __float_as_uint(f);
  return (unsigned short)((u + 0x7fffu + ((u >> 16) & 1u)) >> 16);
}

__global__ void k_cvt(unsigned short* out, const float* in, long n) {
  GS(i, n) out[i] = f2bfu(in[i]);
}

// ---------------------------------------------------------------------
// GEMM: C[M,N] = act(Abf[M,K] @ Wbf[N,K]^T + bias), batched over
// blockIdx.z (element plane strides sA/sC; W, bias shared).
// Block = 256 thr = 8 waves. Tile BM=128 (8 wave strips of 16), BN=64,
// BK=32. W tile async-copied to LDS (bf16, 4KB, one b128 per thread);
// A fragment = two b128 loads in the documented 16-bit A layout.
// ACT: 0=none 1=sigmoid 2=leaky_relu(0.01)
// ---------------------------------------------------------------------
template <int ACT>
__global__ __launch_bounds__(256) void k_gemm(
    const unsigned short* __restrict__ A, const unsigned short* __restrict__ W,
    const float* __restrict__ bias, float* __restrict__ C, int M, int N,
    int K, long sA, long sC) {
  A += (long)blockIdx.z * sA;
  C += (long)blockIdx.z * sC;
  const int wave = threadIdx.x >> 5;
  const int lane = threadIdx.x & 31;
  const int g = lane >> 4;     // lane group (K-half select)
  const int mr = lane & 15;    // row-in-tile / col-in-tile
  const int m0 = blockIdx.x * 128 + wave * 16;
  const int n0 = blockIdx.y * 64;

  __shared__ __align__(16) unsigned short sB[64][40];  // 64 x 32 bf16 (+pad)

  fx8 acc[4] = {};
  const int arow = (m0 + mr < M) ? (m0 + mr) : (M - 1);  // clamp M tail
  const unsigned short* Arow = A + (long)arow * K;
  const bool kAligned = ((K & 7) == 0);

  // staging coords: thread t covers sB[t>>2][(t&3)*8 .. +7] (16 bytes)
  const int sr = threadIdx.x >> 2;
  const int sc = (threadIdx.x & 3) * 8;
  const unsigned ldsAddr = (unsigned)(uintptr_t)(&sB[sr][sc]);

  for (int kk = 0; kk < K; kk += 32) {
    const bool fullK = kAligned && (kk + 32 <= K);
    const bool fullW = fullK && (n0 + 64 <= N);

    // ---- stage W[n0..n0+63, kk..kk+31] -> LDS ----
    if (fullW) {
      const unsigned short* gp = W + (long)(n0 + sr) * K + kk + sc;
      unsigned long long ga = (unsigned long long)(uintptr_t)gp;
      asm volatile("global_load_async_to_lds_b128 %0, %1, off"
                   :: "v"(ldsAddr), "v"(ga) : "memory");
    } else {
      const int wr = n0 + sr;
      const int wrc = (wr < N) ? wr : (N - 1);
#pragma unroll
      for (int i = 0; i < 8; ++i) {
        int kc = kk + sc + i;
        int kcc = (kc < K) ? kc : (K - 1);
        unsigned short v = W[(long)wrc * K + kcc];  // clamped: always in-bounds
        sB[sr][sc + i] = (wr < N && kc < K) ? v : (unsigned short)0;
      }
    }

    // ---- A fragment: elem e -> k = kk + (e/8)*16 + g*8 + e%8 ----
    union { unsigned short s[16]; uint4 u[2]; bfx16 v; } ta;
    if (fullK) {
      ta.u[0] = *(const uint4*)(Arow + kk + g * 8);
      ta.u[1] = *(const uint4*)(Arow + kk + 16 + g * 8);
    } else {
#pragma unroll
      for (int e = 0; e < 8; ++e) {
        int k0 = kk + g * 8 + e, k1 = kk + 16 + g * 8 + e;
        int k0c = (k0 < K) ? k0 : (K - 1);
        int k1c = (k1 < K) ? k1 : (K - 1);
        unsigned short a0 = Arow[k0c];  // clamped loads, no exec-mask branching
        unsigned short a1 = Arow[k1c];
        ta.s[e] = (k0 < K) ? a0 : (unsigned short)0;
        ta.s[e + 8] = (k1 < K) ? a1 : (unsigned short)0;
      }
    }
    if (kk + 32 < K) __builtin_prefetch(Arow + kk + 32, 0, 1);

    if (fullW) asm volatile("s_wait_asynccnt 0" ::: "memory");
    __syncthreads();

    // ---- load all 4 B fragments, then 4 back-to-back WMMAs ----
    bfx16 bt[4];
#pragma unroll
    for (int j = 0; j < 4; ++j) {
      union { uint4 u[2]; bfx16 v; } tb;
      const uint4* prow = (const uint4*)(&sB[j * 16 + mr][0]);
      tb.u[0] = prow[g];       // k = g*8 .. g*8+7
      tb.u[1] = prow[2 + g];   // k = 16+g*8 ..
      bt[j] = tb.v;
    }
#pragma unroll
    for (int j = 0; j < 4; ++j)
      acc[j] = __builtin_amdgcn_wmma_f32_16x16x32_bf16(
          false, ta.v, false, bt[j], (short)0, acc[j], false, false);
    __syncthreads();
  }

  // ---- epilogue: D layout VGPR v -> m = m0 + g*8 + v, n = n0+j*16+mr ----
#pragma unroll
  for (int j = 0; j < 4; ++j) {
    int n = n0 + j * 16 + mr;
    if (n < N) {
      float bb = bias ? bias[n] : 0.f;
#pragma unroll
      for (int v = 0; v < 8; ++v) {
        int m = m0 + g * 8 + v;
        if (m < M) {
          float x = acc[j][v] + bb;
          if (ACT == 1) x = 1.f / (1.f + __expf(-x));
          if (ACT == 2) x = (x >= 0.f) ? x : 0.01f * x;
          C[(long)m * N + n] = x;
        }
      }
    }
  }
}

// ------------------------- elementwise kernels -----------------------
__global__ void k_fill(float* p, long n, float v) { GS(i, n) p[i] = v; }

__global__ void k_add(float* d, const float* s, long n) { GS(i, n) d[i] += s[i]; }

__global__ void k_mul(float* d, const float* s, long n) { GS(i, n) d[i] *= s[i]; }

__global__ void k_leaky(float* x, long n) {
  GS(i, n) { float v = x[i]; x[i] = (v >= 0.f) ? v : 0.01f * v; }
}

// (N,C,3) interleaved -> 3 planar (N,C)
__global__ void k_deint3(float* out, long ps, const float* in, long n) {
  GS(i, n) {
    out[i] = in[3 * i];
    out[ps + i] = in[3 * i + 1];
    out[2 * ps + i] = in[3 * i + 2];
  }
}

// out = sqrt|sum of squares across 3 planes
__global__ void k_norm3(float* out, const float* x, long ps, long n, int dosqrt) {
  GS(i, n) {
    float a = x[i], b = x[ps + i], c = x[2 * ps + i];
    float s = a * a + b * b + c * c;
    out[i] = dosqrt ? sqrtf(s) : s;
  }
}

__global__ void k_gate(float* v, long ps, const float* g, long n) {
  GS(i, n) {
    float s = 1.f / (1.f + __expf(-g[i]));
    v[i] *= s;
    v[ps + i] *= s;
    v[2 * ps + i] *= s;
  }
}

// VN-LeakyReLU: x,d are 3-plane; in-place on x
__global__ void k_vnleaky(float* x, const float* d, long ps, long n) {
  GS(i, n) {
    float xs[3], ds[3], dot = 0.f, dsq = 0.f;
#pragma unroll
    for (int k = 0; k < 3; ++k) {
      xs[k] = x[k * ps + i];
      ds[k] = d[k * ps + i];
      dot += xs[k] * ds[k];
      dsq += ds[k] * ds[k];
    }
    float coef = (dot >= 0.f) ? 0.f : dot / (dsq + 1e-6f);
#pragma unroll
    for (int k = 0; k < 3; ++k)
      x[k * ps + i] = 0.01f * xs[k] + 0.99f * (xs[k] - coef * ds[k]);
  }
}

// vec = pa[ia] - pb[ib] (pb/idx nullable), planar out + dist
__global__ void k_edge_geom(const float* pa, const float* pb, const int* ia,
                            const int* ib, long E, float* vec, long ps,
                            float* dist) {
  GS(i, E) {
    long a = ia ? (long)ia[i] : i;
    float x = pa[3 * a], y = pa[3 * a + 1], z = pa[3 * a + 2];
    if (pb) {
      long b = ib ? (long)ib[i] : i;
      x -= pb[3 * b]; y -= pb[3 * b + 1]; z -= pb[3 * b + 2];
    }
    vec[i] = x; vec[ps + i] = y; vec[2 * ps + i] = z;
    dist[i] = sqrtf(x * x + y * y + z * z);
  }
}

__global__ void k_smear(float* out, const float* dist, long E, int G,
                        float stop, float coeff) {
  GS(t, E * (long)G) {
    long e = t / G; int g = (int)(t % G);
    float off = stop * (float)g / (float)(G - 1);
    float d = dist[e] - off;
    out[t] = __expf(coeff * d * d);
  }
}

// out[d][e,c] = (vec_d/(|v|+1e-7)) * w[c]
__global__ void k_edgeexp(float* out, long ops, const float* vec, long vps,
                          const float* dist, const float* w, long E, int C) {
  GS(t, E * (long)C) {
    long e = t / C; int c = (int)(t % C);
    float inv = 1.f / (dist[e] + 1e-7f);
    float wc = w[c];
    out[t] = vec[e] * inv * wc;
    out[ops + t] = vec[vps + e] * inv * wc;
    out[2 * ops + t] = vec[2 * vps + e] * inv * wc;
  }
}

// gather/concat: dst[p][r][doff+c] = src[p][idx(r)][c]
__global__ void k_gather(float* dst, int dstride, int doff, long dps,
                         const float* src, int sstride, long sps,
                         const int* idx, long rows, int cols, int planes) {
  GS(t, (long)planes * rows * cols) {
    long pc = t / (rows * (long)cols);
    long r2 = t % (rows * (long)cols);
    long r = r2 / cols; int c = (int)(r2 % cols);
    long sr = idx ? (long)idx[r] : r;
    dst[pc * dps + r * dstride + doff + c] = src[pc * sps + sr * sstride + c];
  }
}

__global__ void k_scatter_add(float* dst, long dps, const float* src, long sps,
                              const int* idx, long rows, int cols, int planes) {
  GS(t, (long)planes * rows * cols) {
    long pc = t / (rows * (long)cols);
    long r2 = t % (rows * (long)cols);
    long r = r2 / cols; int c = (int)(r2 % cols);
    atomicAdd(&dst[pc * dps + (long)idx[r] * cols + c],
              src[pc * sps + r * cols + c]);
  }
}

// y[d] = t2*nv[d] + t3*t4[d], planes contiguous
__global__ void k_fma2(float* y, const float* t2, const float* nv,
                       const float* t3, const float* t4, long ps, long total) {
  GS(t, total) {
    long i = t % ps;
    y[t] = t2[i] * nv[t] + t3[i] * t4[t];
  }
}

__global__ void k_cutoff(float* buf, const float* dist, long rows, int cols,
                         int planes, long ps, float cut) {
  GS(t, (long)planes * rows * cols) {
    long pc = t / (rows * (long)cols);
    long r2 = t % (rows * (long)cols);
    long r = r2 / cols;
    float d = dist[r];
    float Cv = (d <= cut && d >= 0.f) ? 0.5f * (__cosf(d * PI_F / cut) + 1.f) : 0.f;
    buf[pc * ps + r2] *= Cv;
  }
}

// ----------------------- attention kernels ---------------------------
__device__ inline void atomicMaxF(float* addr, float val) {
  unsigned* ua = (unsigned*)addr;
  unsigned old = *ua;
  while (__uint_as_float(old) < val) {
    unsigned assumed = old;
    old = atomicCAS(ua, assumed, __float_as_uint(val));
    if (old == assumed) break;
  }
}

__global__ void k_qk(float* ls, float* lv, const float* qs, const float* ks,
                     const float* qv, const float* kv, long vps,
                     const float* bs, const float* bv, const int* pi,
                     const int* pj, long P, int H, int cs, int cv) {
  GS(t, P * (long)H) {
    long p = t / H; int h = (int)(t % H);
    long a = pi[p], b = pj[p];
    const float* qp = qs + a * (long)(H * cs) + h * cs;
    const float* kp = ks + b * (long)(H * cs) + h * cs;
    float ssum = 0.f;
    for (int c = 0; c < cs; ++c) ssum += qp[c] * kp[c];
    float vsum = 0.f;
    for (int d = 0; d < 3; ++d) {
      const float* qq = qv + d * vps + a * (long)(H * cv) + h * cv;
      const float* kk2 = kv + d * vps + b * (long)(H * cv) + h * cv;
      for (int c = 0; c < cv; ++c) vsum += qq[c] * kk2[c];
    }
    ls[t] = bs[t] + ssum;
    lv[t] = bv[t] + vsum;
  }
}

__global__ void k_segmax(float* m, const float* l, const int* idx, long P, int H) {
  GS(t, P * (long)H) atomicMaxF(&m[(long)idx[t / H] * H + t % H], l[t]);
}
__global__ void k_expsum(float* ex, float* s, const float* l, const float* m,
                         const int* idx, long P, int H) {
  GS(t, P * (long)H) {
    long r = (long)idx[t / H] * H + t % H;
    float e = __expf(l[t] - m[r]);
    ex[t] = e;
    atomicAdd(&s[r], e);
  }
}
__global__ void k_divs(float* ex, const float* s, const int* idx, long P, int H) {
  GS(t, P * (long)H) ex[t] /= s[(long)idx[t / H] * H + t % H];
}

__global__ void k_attn(float* o, long ops, const float* v, long vps,
                       const float* al, const int* pi, const int* pj, long P,
                       int cols, int planes, int cph) {
  GS(t, (long)planes * P * cols) {
    long pl = t / (P * (long)cols);
    long r2 = t % (P * (long)cols);
    long p = r2 / cols; int c = (int)(r2 % cols);
    int h = c / cph;
    float a = al[p * (long)(cols / cph) + h];
    atomicAdd(&o[pl * ops + (long)pi[p] * cols + c],
              a * v[pl * vps + (long)pj[p] * cols + c]);
  }
}

__global__ void k_ln_s(float* x, const float* g, const float* b, long R, int C) {
  GS(r, R) {
    float m = 0.f;
    for (int c = 0; c < C; ++c) m += x[r * C + c];
    m /= C;
    float v = 0.f;
    for (int c = 0; c < C; ++c) { float d = x[r * C + c] - m; v += d * d; }
    v /= C;
    float inv = rsqrtf(v + 1e-5f);
    for (int c = 0; c < C; ++c)
      x[r * C + c] = (x[r * C + c] - m) * inv * g[c] + b[c];
  }
}

// g,b stored (C,3) interleaved
__global__ void k_ln_v(float* x, long ps, const float* g, const float* b,
                       long R, int C) {
  GS(r, R) {
    float m = 0.f;
    for (int d = 0; d < 3; ++d)
      for (int c = 0; c < C; ++c) m += x[d * ps + r * C + c];
    m /= (3.f * C);
    float v = 0.f;
    for (int d = 0; d < 3; ++d)
      for (int c = 0; c < C; ++c) {
        float t = x[d * ps + r * C + c] - m;
        v += t * t;
      }
    v /= (3.f * C);
    float inv = rsqrtf(v + 1e-5f);
    for (int d = 0; d < 3; ++d)
      for (int c = 0; c < C; ++c)
        x[d * ps + r * C + c] =
            (x[d * ps + r * C + c] - m) * inv * g[c * 3 + d] + b[c * 3 + d];
  }
}

// ========================= host orchestration ========================
namespace {

struct Arena {
  char* base; size_t off;
  float* f(long n) {
    size_t need = (((size_t)n * 4) + 255) & ~(size_t)255;
    float* p = (float*)(base + off);
    off += need;
    return p;
  }
  unsigned short* h(long n) {
    size_t need = (((size_t)n * 2) + 255) & ~(size_t)255;
    unsigned short* p = (unsigned short*)(base + off);
    off += need;
    return p;
  }
};

struct GBW { const float *bg, *wg, *ws, *wv1, *wv2; };

static GBW gbw(void* const* din, int i) {
  return {(const float*)din[i], (const float*)din[i + 1],
          (const float*)din[i + 2], (const float*)din[i + 3],
          (const float*)din[i + 4]};
}

static inline dim3 gsz(long n) {
  long b = (n + 255) / 256;
  if (b < 1) b = 1;
  if (b > 262144) b = 262144;
  return dim3((unsigned)b);
}

static void gemm(hipStream_t st, Arena& ar, const float* A, const float* W,
                 const float* bias, float* C, long M, int N, int K, int planes,
                 long sA, long sC, int act) {
  // pre-convert A (planes assumed contiguous with stride sA) and W to bf16
  long atot = (long)(planes - 1) * sA + M * (long)K;
  unsigned short* Ab = ar.h(atot);
  k_cvt<<<gsz(atot), 256, 0, st>>>(Ab, A, atot);
  long wtot = (long)N * K;
  unsigned short* Wb = ar.h(wtot);
  k_cvt<<<gsz(wtot), 256, 0, st>>>(Wb, W, wtot);

  dim3 grid((unsigned)((M + 127) / 128), (unsigned)((N + 63) / 64),
            (unsigned)planes);
  dim3 blk(256);
  switch (act) {
    case 1: k_gemm<1><<<grid, blk, 0, st>>>(Ab, Wb, bias, C, (int)M, N, K, sA, sC); break;
    case 2: k_gemm<2><<<grid, blk, 0, st>>>(Ab, Wb, bias, C, (int)M, N, K, sA, sC); break;
    default: k_gemm<0><<<grid, blk, 0, st>>>(Ab, Wb, bias, C, (int)M, N, K, sA, sC); break;
  }
}

// GBLinear: sca(N,in_s) + vec planes(3,N,in_v) -> out_sca(N,out_s),
// out_vec planes(3,N,out_v)
static void gb_linear(hipStream_t st, Arena& ar, const GBW& w,
                      const float* sca, const float* vec, long N, int in_s,
                      int in_v, int out_s, int out_v, float* out_sca,
                      float* out_vec) {
  const int hid = in_v > out_v ? in_v : out_v;
  float* vi = ar.f(3L * N * hid);
  gemm(st, ar, vec, w.wv1, nullptr, vi, N, hid, in_v, 3, N * (long)in_v,
       N * (long)hid, 0);
  float* vn = ar.f(N * (long)hid);
  k_norm3<<<gsz(N * (long)hid), 256, 0, st>>>(vn, vi, N * (long)hid,
                                              N * (long)hid, 1);
  const int cc = hid + in_s;
  float* cat = ar.f(N * (long)cc);
  k_gather<<<gsz(N * (long)hid), 256, 0, st>>>(cat, cc, 0, 0, vn, hid, 0,
                                               nullptr, N, hid, 1);
  k_gather<<<gsz(N * (long)in_s), 256, 0, st>>>(cat, cc, hid, 0, sca, in_s, 0,
                                                nullptr, N, in_s, 1);
  gemm(st, ar, cat, w.ws, nullptr, out_sca, N, out_s, cc, 1, 0, 0, 0);
  gemm(st, ar, vi, w.wv2, nullptr, out_vec, N, out_v, hid, 3, N * (long)hid,
       N * (long)out_v, 0);
  float* gl = ar.f(N * (long)out_v);
  gemm(st, ar, out_sca, w.wg, w.bg, gl, N, out_v, out_s, 1, 0, 0, 0);
  k_gate<<<gsz(N * (long)out_v), 256, 0, st>>>(out_vec, N * (long)out_v, gl,
                                               N * (long)out_v);
}

static void gb_perceptron(hipStream_t st, Arena& ar, const GBW& w,
                          const float* vn_w, const float* sca,
                          const float* vec, long N, int in_s, int in_v,
                          int out_s, int out_v, float* out_sca,
                          float* out_vec) {
  gb_linear(st, ar, w, sca, vec, N, in_s, in_v, out_s, out_v, out_sca, out_vec);
  float* dd = ar.f(3L * N * out_v);
  gemm(st, ar, out_vec, vn_w, nullptr, dd, N, out_v, out_v, 3,
       N * (long)out_v, N * (long)out_v, 0);
  k_vnleaky<<<gsz(N * (long)out_v), 256, 0, st>>>(out_vec, dd,
                                                  N * (long)out_v,
                                                  N * (long)out_v);
  k_leaky<<<gsz(N * (long)out_s), 256, 0, st>>>(out_sca, N * (long)out_s);
}

static void scatter_softmax(hipStream_t st, Arena& ar, const float* logits,
                            const int* idx, long P, int H, long NSEG,
                            float* out) {
  size_t mk = ar.off;
  float* m = ar.f(NSEG * H);
  float* s = ar.f(NSEG * H);
  k_fill<<<gsz(NSEG * H), 256, 0, st>>>(m, NSEG * (long)H, -1e30f);
  k_fill<<<gsz(NSEG * H), 256, 0, st>>>(s, NSEG * (long)H, 0.f);
  k_segmax<<<gsz(P * H), 256, 0, st>>>(m, logits, idx, P, H);
  k_expsum<<<gsz(P * H), 256, 0, st>>>(out, s, logits, m, idx, P, H);
  k_divs<<<gsz(P * H), 256, 0, st>>>(out, s, idx, P, H);
  ar.off = mk;
}

}  // namespace

extern "C" void kernel_launch(void* const* d_in, const int* in_sizes, int n_in,
                              void* d_out, int out_size, void* d_ws,
                              size_t ws_size, hipStream_t stream) {
  (void)n_in; (void)out_size; (void)ws_size;
  hipStream_t st = stream;

  // ---- dims ----
  const long NQ = in_sizes[0] / 3;
  const long EQ = in_sizes[1] / 2;
  const long NC = in_sizes[2] / 3;
  const long EK = in_sizes[5] / 2;
  const long P  = in_sizes[6] / 2;

  // ---- raw inputs ----
  const float* pos_query = (const float*)d_in[0];
  const int*   eiq       = (const int*)d_in[1];
  const float* cpx_pos   = (const float*)d_in[2];
  const float* node_sca  = (const float*)d_in[3];
  const float* node_vec  = (const float*)d_in[4];
  const int*   eik       = (const int*)d_in[5];
  const int*   pidx      = (const int*)d_in[6];
  const int*   tei       = (const int*)d_in[7];
  const float* tef       = (const float*)d_in[8];
  const float* atom_emb  = (const float*)d_in[9];

  // ---- params (jax pytree dict: sorted-key flatten order) ----
  GBW atten_bias = gbw(d_in, 10);
  const float* bias_vec_w = (const float*)d_in[15];
  GBW atten_k = gbw(d_in, 16);
  const float* ln_s_b = (const float*)d_in[21];
  const float* ln_s_g = (const float*)d_in[22];
  const float* ln_v_b = (const float*)d_in[23];
  const float* ln_v_g = (const float*)d_in[24];
  GBW atten_q = gbw(d_in, 25);
  GBW atten_v = gbw(d_in, 30);
  GBW ef_gb = gbw(d_in, 35);
  GBW ef_gbp_gb = gbw(d_in, 40);
  const float* ef_gbp_vn = (const float*)d_in[45];
  const float* e2n_b = (const float*)d_in[46];
  const float* e2n_w = (const float*)d_in[47];
  GBW edge_gbp_gb = gbw(d_in, 48);
  const float* edge_gbp_vn = (const float*)d_in[53];
  const float* edge_vn_w = (const float*)d_in[54];
  const float* n2e_b = (const float*)d_in[55];
  const float* n2e_w = (const float*)d_in[56];
  GBW node_gb = gbw(d_in, 57);
  GBW out_gb = gbw(d_in, 62);
  const float* sca_b = (const float*)d_in[67];
  const float* sca_w = (const float*)d_in[68];
  GBW nne_gb = gbw(d_in, 69);
  GBW nne_gbp_gb = gbw(d_in, 74);
  const float* nne_gbp_vn = (const float*)d_in[79];
  GBW pred = gbw(d_in, 80);
  GBW root_gb = gbw(d_in, 85);
  const float* root_vec_w = (const float*)d_in[90];
  const float* vec_exp_w = (const float*)d_in[91];

  Arena ar{(char*)d_ws, 0};

  // ---- persistent buffers ----
  float* y0  = ar.f(NQ * 128);       // node scalar features
  float* y1  = ar.f(3L * NQ * 32);   // node vector planes
  float* a_s = ar.f(EQ * 128);       // edge features
  float* a_v = ar.f(3L * EQ * 32);
  float* o_s = ar.f(EQ * 128);       // attention out
  float* o_v = ar.f(3L * EQ * 32);
  float* nvp = ar.f(3L * NC * 64);   // node_vec deinterleaved
  const size_t mark0 = ar.off;

  k_deint3<<<gsz(NC * 64), 256, 0, st>>>(nvp, NC * 64L, node_vec, NC * 64L);

  // ================= Stage A: kNN message passing ====================
  {
    const int* rowk = eik;
    const int* colk = eik + EK;
    float* vq = ar.f(3 * EK);
    float* dq = ar.f(EK);
    k_edge_geom<<<gsz(EK), 256, 0, st>>>(pos_query, cpx_pos, rowk, colk, EK,
                                         vq, EK, dq);
    float* esca = ar.f(EK * 64);
    {
      float step = 10.f / 63.f;
      k_smear<<<gsz(EK * 64), 256, 0, st>>>(esca, dq, EK, 64, 10.f,
                                            -0.5f / (step * step));
    }
    float* evec = ar.f(3L * EK * 64);
    k_edgeexp<<<gsz(EK * 64), 256, 0, st>>>(evec, EK * 64L, vq, EK, dq,
                                            vec_exp_w, EK, 64);
    float* ns_node = ar.f(NC * 128);
    float* nv_node = ar.f(3L * NC * 32);
    gb_linear(st, ar, node_gb, node_sca, nvp, NC, 256, 64, 128, 32, ns_node,
              nv_node);
    float* n_s = ar.f(EK * 128);
    float* n_v = ar.f(3L * EK * 32);
    k_gather<<<gsz(EK * 128), 256, 0, st>>>(n_s, 128, 0, 0, ns_node, 128, 0,
                                            colk, EK, 128, 1);
    k_gather<<<gsz(3L * EK * 32), 256, 0, st>>>(n_v, 32, 0, EK * 32L, nv_node,
                                                32, NC * 32L, colk, EK, 32, 3);
    float* es = ar.f(EK * 64);
    float* ev = ar.f(3L * EK * 64);
    gb_perceptron(st, ar, edge_gbp_gb, edge_gbp_vn, esca, evec, EK, 64, 64, 64,
                  64, es, ev);
    float* t3 = ar.f(EK * 32);
    gemm(st, ar, n_s, n2e_w, n2e_b, t3, EK, 32, 128, 1, 0, 0, 0);
    float* t1 = ar.f(EK * 128);
    gemm(st, ar, es, sca_w, sca_b, t1, EK, 128, 64, 1, 0, 0, 0);
    k_mul<<<gsz(EK * 128), 256, 0, st>>>(n_s, t1, EK * 128L);  // y_sca in n_s
    float* t2 = ar.f(EK * 32);
    gemm(st, ar, es, e2n_w, e2n_b, t2, EK, 32, 64, 1, 0, 0, 0);
    float* t4 = ar.f(3L * EK * 32);
    gemm(st, ar, ev, edge_vn_w, nullptr, t4, EK, 32, 64, 3, EK * 64L,
         EK * 32L, 0);
    float* yv = ar.f(3L * EK * 32);
    k_fma2<<<gsz(3L * EK * 32), 256, 0, st>>>(yv, t2, n_v, t3, t4, EK * 32L,
                                              3L * EK * 32);
    float* hs = ar.f(EK * 128);
    float* hv = ar.f(3L * EK * 32);
    gb_linear(st, ar, out_gb, n_s, yv, EK, 128, 32, 128, 32, hs, hv);
    k_cutoff<<<gsz(EK * 128), 256, 0, st>>>(hs, dq, EK, 128, 1, 0L, 10.f);
    k_cutoff<<<gsz(3L * EK * 32), 256, 0, st>>>(hv, dq, EK, 32, 3, EK * 32L,
                                                10.f);
    k_fill<<<gsz(NQ * 128), 256, 0, st>>>(y0, NQ * 128L, 0.f);
    k_fill<<<gsz(3L * NQ * 32), 256, 0, st>>>(y1, 3L * NQ * 32, 0.f);
    k_scatter_add<<<gsz(EK * 128), 256, 0, st>>>(y0, 0L, hs, 0L, rowk, EK, 128,
                                                 1);
    k_scatter_add<<<gsz(3L * EK * 32), 256, 0, st>>>(y1, NQ * 32L, hv,
                                                     EK * 32L, rowk, EK, 32, 3);
    ar.off = mark0;
  }

  // ================= Stage B: root term ==============================
  {
    float* vp = ar.f(3 * NQ);
    float* dp = ar.f(NQ);
    k_edge_geom<<<gsz(NQ), 256, 0, st>>>(pos_query, nullptr, nullptr, nullptr,
                                         NQ, vp, NQ, dp);
    float* rv = ar.f(3L * NQ * 64);
    k_edgeexp<<<gsz(NQ * 64), 256, 0, st>>>(rv, NQ * 64L, vp, NQ, dp,
                                            root_vec_w, NQ, 64);
    float* rs = ar.f(NQ * 128);
    float* rvv = ar.f(3L * NQ * 32);
    gb_linear(st, ar, root_gb, atom_emb, rv, NQ, 256, 64, 128, 32, rs, rvv);
    k_add<<<gsz(NQ * 128), 256, 0, st>>>(y0, rs, NQ * 128L);
    k_add<<<gsz(3L * NQ * 32), 256, 0, st>>>(y1, rvv, 3L * NQ * 32);
    ar.off = mark0;
  }

  // ================= Stage C: candidate-edge features ================
  {
    const int* ii = eiq;
    const int* jj = eiq + EQ;
    float* vq = ar.f(3 * EQ);
    float* dq = ar.f(EQ);
    k_edge_geom<<<gsz(EQ), 256, 0, st>>>(pos_query, cpx_pos, ii, jj, EQ, vq,
                                         EQ, dq);
    float* eqs = ar.f(EQ * 64);
    {
      float step = 3.f / 63.f;
      k_smear<<<gsz(EQ * 64), 256, 0, st>>>(eqs, dq, EQ, 64, 3.f,
                                            -0.5f / (step * step));
    }
    float* eqv = ar.f(3L * EQ * 64);
    k_edgeexp<<<gsz(EQ * 64), 256, 0, st>>>(eqv, EQ * 64L, vq, EQ, dq,
                                            vec_exp_w, EQ, 64);
    float* fs = ar.f(EQ * 128);
    float* fv = ar.f(3L * EQ * 32);
    gb_perceptron(st, ar, nne_gbp_gb, nne_gbp_vn, eqs, eqv, EQ, 64, 64, 128,
                  32, fs, fv);
    float* fs2 = ar.f(EQ * 128);
    float* fv2 = ar.f(3L * EQ * 32);
    gb_linear(st, ar, nne_gb, fs, fv, EQ, 128, 32, 128, 32, fs2, fv2);
    float* cs = ar.f(EQ * 512);
    k_gather<<<gsz(EQ * 128), 256, 0, st>>>(cs, 512, 0, 0, y0, 128, 0, ii, EQ,
                                            128, 1);
    k_gather<<<gsz(EQ * 256), 256, 0, st>>>(cs, 512, 128, 0, node_sca, 256, 0,
                                            jj, EQ, 256, 1);
    k_gather<<<gsz(EQ * 128), 256, 0, st>>>(cs, 512, 384, 0, fs2, 128, 0,
                                            nullptr, EQ, 128, 1);
    float* cv = ar.f(3L * EQ * 128);
    k_gather<<<gsz(3L * EQ * 32), 256, 0, st>>>(cv, 128, 0, EQ * 128L, y1, 32,
                                                NQ * 32L, ii, EQ, 32, 3);
    k_gather<<<gsz(3L * EQ * 64), 256, 0, st>>>(cv, 128, 32, EQ * 128L, nvp,
                                                64, NC * 64L, jj, EQ, 64, 3);
    k_gather<<<gsz(3L * EQ * 32), 256, 0, st>>>(cv, 128, 96, EQ * 128L, fv2,
                                                32, EQ * 32L, nullptr, EQ, 32,
                                                3);
    float* xs = ar.f(EQ * 128);
    float* xv = ar.f(3L * EQ * 32);
    gb_perceptron(st, ar, ef_gbp_gb, ef_gbp_vn, cs, cv, EQ, 512, 128, 128, 32,
                  xs, xv);
    gb_linear(st, ar, ef_gb, xs, xv, EQ, 128, 32, 128, 32, a_s, a_v);
    ar.off = mark0;
  }

  // ================= Stage D: triangle-aware attention ===============
  {
    float* qs = ar.f(EQ * 128); float* qv = ar.f(3L * EQ * 32);
    float* ks2 = ar.f(EQ * 128); float* kv = ar.f(3L * EQ * 32);
    float* vs = ar.f(EQ * 128); float* vv = ar.f(3L * EQ * 32);
    gb_linear(st, ar, atten_q, a_s, a_v, EQ, 128, 32, 128, 32, qs, qv);
    gb_linear(st, ar, atten_k, a_s, a_v, EQ, 128, 32, 128, 32, ks2, kv);
    gb_linear(st, ar, atten_v, a_s, a_v, EQ, 128, 32, 128, 32, vs, vv);

    const int* na = tei;
    const int* nb = tei + P;
    float* tv = ar.f(3 * P);
    float* td = ar.f(P);
    k_edge_geom<<<gsz(P), 256, 0, st>>>(cpx_pos, cpx_pos, na, nb, P, tv, P, td);
    float* tds = ar.f(P * 124);
    {
      float step = 10.f / 123.f;
      k_smear<<<gsz(P * 124), 256, 0, st>>>(tds, td, P, 124, 10.f,
                                            -0.5f / (step * step));
    }
    float* tcat = ar.f(P * 128);
    k_gather<<<gsz(P * 124), 256, 0, st>>>(tcat, 128, 0, 0, tds, 124, 0,
                                           nullptr, P, 124, 1);
    k_gather<<<gsz(P * 4), 256, 0, st>>>(tcat, 128, 124, 0, tef, 4, 0, nullptr,
                                         P, 4, 1);
    float* tvp = ar.f(3L * P * 32);
    k_edgeexp<<<gsz(P * 32), 256, 0, st>>>(tvp, P * 32L, tv, P, td, bias_vec_w,
                                           P, 32);
    float* bs = ar.f(P * 4);
    float* bvv = ar.f(3L * P * 4);
    gb_linear(st, ar, atten_bias, tcat, tvp, P, 128, 32, 4, 4, bs, bvv);
    float* bv = ar.f(P * 4);
    k_norm3<<<gsz(P * 4), 256, 0, st>>>(bv, bvv, P * 4L, P * 4L, 0);

    const int* pi = pidx;
    const int* pj = pidx + P;
    float* ls = ar.f(P * 4);
    float* lv = ar.f(P * 4);
    k_qk<<<gsz(P * 4), 256, 0, st>>>(ls, lv, qs, ks2, qv, kv, EQ * 32L, bs, bv,
                                     pi, pj, P, 4, 32, 8);
    float* als = ar.f(P * 4);
    float* alv = ar.f(P * 4);
    scatter_softmax(st, ar, ls, pi, P, 4, EQ, als);
    scatter_softmax(st, ar, lv, pi, P, 4, EQ, alv);

    k_fill<<<gsz(EQ * 128), 256, 0, st>>>(o_s, EQ * 128L, 0.f);
    k_fill<<<gsz(3L * EQ * 32), 256, 0, st>>>(o_v, 3L * EQ * 32, 0.f);
    k_attn<<<gsz(P * 128), 256, 0, st>>>(o_s, EQ * 128L, vs, EQ * 128L, als,
                                         pi, pj, P, 128, 1, 32);
    k_attn<<<gsz(3L * P * 32), 256, 0, st>>>(o_v, EQ * 32L, vv, EQ * 32L, alv,
                                             pi, pj, P, 32, 3, 8);
    k_add<<<gsz(EQ * 128), 256, 0, st>>>(o_s, a_s, EQ * 128L);
    k_add<<<gsz(3L * EQ * 32), 256, 0, st>>>(o_v, a_v, 3L * EQ * 32);
    k_ln_s<<<gsz(EQ), 256, 0, st>>>(o_s, ln_s_g, ln_s_b, EQ, 128);
    k_ln_v<<<gsz(EQ), 256, 0, st>>>(o_v, EQ * 32L, ln_v_g, ln_v_b, EQ, 32);
    ar.off = mark0;
  }

  // ================= Stage E: prediction head ========================
  {
    float* vi = ar.f(3L * EQ * 32);
    gemm(st, ar, o_v, pred.wv1, nullptr, vi, EQ, 32, 32, 3, EQ * 32L,
         EQ * 32L, 0);
    float* vn = ar.f(EQ * 32);
    k_norm3<<<gsz(EQ * 32), 256, 0, st>>>(vn, vi, EQ * 32L, EQ * 32L, 1);
    float* cat = ar.f(EQ * 160);
    k_gather<<<gsz(EQ * 32), 256, 0, st>>>(cat, 160, 0, 0, vn, 32, 0, nullptr,
                                           EQ, 32, 1);
    k_gather<<<gsz(EQ * 128), 256, 0, st>>>(cat, 160, 32, 0, o_s, 128, 0,
                                            nullptr, EQ, 128, 1);
    gemm(st, ar, cat, pred.ws, nullptr, (float*)d_out, EQ, 4, 160, 1, 0, 0, 0);
    ar.off = mark0;
  }
}
